// DGCRN_18141941858349
// MI455X (gfx1250) — matrix operand order repaired and checked by reading, not compiled, not run on hardware
//
#include <hip/hip_runtime.h>
#include <hip/hip_bf16.h>
#include <math.h>

// DGCRN forward for MI455X (gfx1250, wave32, WMMA).
// All node-mixing matmuls go through v_wmma_f32_16x16x32_f16 (f16 in, f32 acc).
// B operands (shared by all 8 waves of a block) are staged in LDS. K loops are
// fully unrolled and BOTH A (global) and B (LDS) fragments are double-buffered
// across chunks, so fragment rotation is SSA renaming and every WMMA waits on
// a partial counter value instead of cnt==0.

#define B_    16
#define N_    256
#define T_    12
#define HID_  64
#define CH_   65      // 1 + HID
#define CHP_  80      // channel dim padded to 5 wmma tiles
#define DIN_  195     // (GDEP+1)*CH
#define KCAT_ 480     // 2 * 3 * CHP (fwd|bwd concat, padded)
#define NDP_  64      // NODE_DIM 40 padded to 64

typedef _Float16 v16h __attribute__((ext_vector_type(16)));
typedef _Float16 v8h  __attribute__((ext_vector_type(8)));
typedef float    v8f  __attribute__((ext_vector_type(8)));

// ---------------- WMMA fragment loaders (ISA 7.12.2 layouts) ----------------
// A operand (16x32 f16), row-major src [M][K], ld%8==0, k0%32==0.
// lane<16: row=lane, halves = K[k0..k0+7], K[k0+16..k0+23]; lane>=16: +8 shift.
__device__ __forceinline__ v16h load_frag_a(const _Float16* __restrict__ p, int ld,
                                            int m0, int k0, int lane) {
  int row = m0 + (lane & 15);
  int kb  = k0 + ((lane >> 4) << 3);
  const _Float16* q = p + (size_t)row * ld + kb;
  v8h lo = *(const v8h*)(q);
  v8h hi = *(const v8h*)(q + 16);
  v16h r;
#pragma unroll
  for (int i = 0; i < 8; ++i) { r[i] = lo[i]; r[i + 8] = hi[i]; }
  return r;
}
// B operand (32x16 f16) from K-contiguous ("transposed") src [Ncol][K] in LDS:
// lane L holds col n0+(L&15), K = k0 + 8*(L>>4) + 0..7, then +16 (matches A).
__device__ __forceinline__ v16h load_frag_b_lds(const _Float16* pT, int ld,
                                                int n0, int k0, int lane) {
  int col = n0 + (lane & 15);
  int kb  = k0 + ((lane >> 4) << 3);
  const _Float16* q = pT + (size_t)col * ld + kb;
  v8h lo = *(const v8h*)(q);
  v8h hi = *(const v8h*)(q + 16);
  v16h r;
#pragma unroll
  for (int i = 0; i < 8; ++i) { r[i] = lo[i]; r[i + 8] = hi[i]; }
  return r;
}
__device__ __forceinline__ v8f wmma16(v16h a, v16h b, v8f c) {
  return __builtin_amdgcn_wmma_f32_16x16x32_f16(false, a, false, b, (short)0, c,
                                                false, false);
}

// ---------------- graph conv GEMM: Y[b] = A1-op @ X  (+ A2-op @ X) ----------
// A-op matrices stored [w][v] (pre-transposed), X stored transposed XT[b][c][v].
// Block = 8 waves; wave w -> rows [m0,m0+16) x 80 cols (5 tiles). K = 256.
// XT panel (80x256 f16 = 40KB) staged in LDS once; shared by all waves.
template <bool DUAL>
__global__ __launch_bounds__(256) void k_gemm_graph(
    const _Float16* __restrict__ A1, long sA1,
    const _Float16* __restrict__ A2, long sA2,
    const _Float16* __restrict__ XT, float* __restrict__ Y) {
  __shared__ _Float16 sX[CHP_ * N_];  // [c][v], ld = 256
  int b = blockIdx.y;
  int wave = threadIdx.x >> 5, lane = threadIdx.x & 31;
  int m0 = blockIdx.x * 128 + wave * 16;
  const _Float16* a1 = A1 + (size_t)b * sA1;
  const _Float16* a2 = DUAL ? (A2 + (size_t)b * sA2) : nullptr;
  const _Float16* bt = XT + (size_t)b * CHP_ * N_;
  // cooperative coalesced stage: 20480 halves / 256 threads = 10 x v8h each
#pragma unroll
  for (int it = 0; it < (CHP_ * N_) / (256 * 8); ++it) {
    int idx = (it * 256 + threadIdx.x) * 8;
    *(v8h*)(sX + idx) = *(const v8h*)(bt + idx);
  }
  __syncthreads();

  v8f acc[5] = {};
  // prologue: chunk-0 fragments
  v16h a1c = load_frag_a(a1, N_, m0, 0, lane);
  v16h a2c{};
  if constexpr (DUAL) a2c = load_frag_a(a2, N_, m0, 0, lane);
  v16h bfc[5];
#pragma unroll
  for (int j = 0; j < 5; ++j) bfc[j] = load_frag_b_lds(sX, N_, j * 16, 0, lane);
#pragma unroll
  for (int k0 = 0; k0 < N_; k0 += 32) {
    // issue next-chunk fragment loads before consuming current (full unroll
    // renames the rotation -> no copies, partial s_wait values)
    v16h a1n{}, a2n{};
    v16h bfn[5] = {};
    if (k0 + 32 < N_) {
      a1n = load_frag_a(a1, N_, m0, k0 + 32, lane);
      if constexpr (DUAL) a2n = load_frag_a(a2, N_, m0, k0 + 32, lane);
#pragma unroll
      for (int j = 0; j < 5; ++j)
        bfn[j] = load_frag_b_lds(sX, N_, j * 16, k0 + 32, lane);
    }
#pragma unroll
    for (int j = 0; j < 5; ++j) {
      acc[j] = wmma16(a1c, bfc[j], acc[j]);
      if constexpr (DUAL) acc[j] = wmma16(a2c, bfc[j], acc[j]);
    }
    a1c = a1n;
    if constexpr (DUAL) a2c = a2n;
#pragma unroll
    for (int j = 0; j < 5; ++j) bfc[j] = bfn[j];
  }
  float* yb = Y + (size_t)b * N_ * CHP_;
  int rbase = m0 + ((lane >> 4) << 3);
  int c0 = lane & 15;
#pragma unroll
  for (int j = 0; j < 5; ++j)
#pragma unroll
    for (int r = 0; r < 8; ++r)
      yb[(size_t)(rbase + r) * CHP_ + j * 16 + c0] = acc[j][r];
}

// ---------------- generic GEMM: C = A(MxK) @ BT(NxK)^T, wave = 16x64 --------
// K is a template constant: LDS panel statically sized, K loop fully unrolled.
template <int K>
__global__ __launch_bounds__(256) void k_gemm_nt4(
    const _Float16* __restrict__ A, int lda, long sA,
    const _Float16* __restrict__ BT, int ldb, long sB,
    float* __restrict__ C, int ldc, long sC) {
  __shared__ _Float16 sB64[64 * K];  // [64][K]
  int b = blockIdx.z;
  int wave = threadIdx.x >> 5, lane = threadIdx.x & 31;
  int m0 = blockIdx.x * 128 + wave * 16;
  int n0 = blockIdx.y * 64;
  A += (size_t)b * sA;
  BT += (size_t)b * sB;
  C += (size_t)b * sC;
  // stage BT rows n0..n0+63 -> LDS (ld = K), 16B chunks, coalesced in K
#pragma unroll
  for (int it = 0; it < (64 * K) / (256 * 8); ++it) {
    int idx = (it * 256 + threadIdx.x) * 8;
    int col = idx / K, kk = idx % K;
    *(v8h*)(sB64 + (size_t)col * K + kk) =
        *(const v8h*)(BT + (size_t)(n0 + col) * ldb + kk);
  }
  __syncthreads();

  v8f acc[4] = {};
  v16h ac = load_frag_a(A, lda, m0, 0, lane);
  v16h bfc[4];
#pragma unroll
  for (int j = 0; j < 4; ++j) bfc[j] = load_frag_b_lds(sB64, K, j * 16, 0, lane);
#pragma unroll
  for (int k0 = 0; k0 < K; k0 += 32) {
    v16h an{};
    v16h bfn[4] = {};
    if (k0 + 32 < K) {
      an = load_frag_a(A, lda, m0, k0 + 32, lane);
#pragma unroll
      for (int j = 0; j < 4; ++j)
        bfn[j] = load_frag_b_lds(sB64, K, j * 16, k0 + 32, lane);
    }
#pragma unroll
    for (int j = 0; j < 4; ++j) acc[j] = wmma16(ac, bfc[j], acc[j]);
    ac = an;
#pragma unroll
    for (int j = 0; j < 4; ++j) bfc[j] = bfn[j];
  }
  int rbase = m0 + ((lane >> 4) << 3);
  int c0 = lane & 15;
#pragma unroll
  for (int j = 0; j < 4; ++j)
#pragma unroll
    for (int r = 0; r < 8; ++r)
      C[(size_t)(rbase + r) * ldc + n0 + j * 16 + c0] = acc[j][r];
}

// ---------------- elementwise / packing kernels -----------------------------
__global__ void k_pack(const float* __restrict__ xin, int xstride,
                       const float* __restrict__ H, float* __restrict__ comb,
                       _Float16* __restrict__ XT0, _Float16* __restrict__ conc,
                       float* __restrict__ hoF, float* __restrict__ hoB) {
  int id = blockIdx.x * blockDim.x + threadIdx.x;
  if (id >= B_ * N_) return;
  int n = id & (N_ - 1), b = id >> 8;
  float x = xin[(size_t)id * xstride];
  comb[(size_t)id * CH_] = x;
  hoF[(size_t)id * DIN_] = x;
  hoB[(size_t)id * DIN_] = x;
  conc[(size_t)id * KCAT_] = (_Float16)x;
  conc[(size_t)id * KCAT_ + 240] = (_Float16)x;
  XT0[((size_t)b * CHP_) * N_ + n] = (_Float16)x;
  for (int c = 0; c < HID_; ++c) {
    float hv = H[(size_t)id * HID_ + c];
    comb[(size_t)id * CH_ + 1 + c] = hv;
    hoF[(size_t)id * DIN_ + 1 + c] = hv;
    hoB[(size_t)id * DIN_ + 1 + c] = hv;
    conc[(size_t)id * KCAT_ + 1 + c] = (_Float16)hv;
    conc[(size_t)id * KCAT_ + 241 + c] = (_Float16)hv;
    XT0[((size_t)b * CHP_ + 1 + c) * N_ + n] = (_Float16)hv;
  }
}

// h = 0.05*res + 0.95*Y ; fan out to f32 concat / f16 concat / transposed f16.
__global__ void k_combine(const float* __restrict__ Yg, const float* __restrict__ res,
                          float* __restrict__ ho, int hoOff,
                          _Float16* __restrict__ conc, int concOff,
                          _Float16* __restrict__ XTout) {
  int e = blockIdx.x * blockDim.x + threadIdx.x;
  if (e >= B_ * N_ * CH_) return;
  int c = e % CH_;
  int id = e / CH_;
  int n = id & (N_ - 1), b = id >> 8;
  float h = 0.05f * res[(size_t)id * CH_ + c] + 0.95f * Yg[(size_t)id * CHP_ + c];
  if (ho) ho[(size_t)id * DIN_ + hoOff + c] = h;
  if (conc) conc[(size_t)id * KCAT_ + concOff + c] = (_Float16)h;
  if (XTout) XTout[((size_t)b * CHP_ + c) * N_ + n] = (_Float16)h;
}

struct MlpP { const float* p[4][6]; };  // [h1a,h1b,h2a,h2b][fc1W,fc1b,fc2W,fc2b,fc3W,fc3b]

__device__ __forceinline__ void mlp_eval(const float* __restrict__ ho,
                                         const float* const* w,
                                         float* __restrict__ out40) {
  const float *W1 = w[0], *b1 = w[1], *W2 = w[2], *b2 = w[3], *W3 = w[4], *b3 = w[5];
  float h1[16];
  for (int j = 0; j < 16; ++j) {
    float a = b1[j];
    for (int i = 0; i < DIN_; ++i) a += ho[i] * W1[i * 16 + j];
    h1[j] = 1.f / (1.f + __expf(-a));
  }
  float h2[2];
  for (int j = 0; j < 2; ++j) {
    float a = b2[j];
    for (int i = 0; i < 16; ++i) a += h1[i] * W2[i * 2 + j];
    h2[j] = 1.f / (1.f + __expf(-a));
  }
  for (int j = 0; j < 40; ++j)
    out40[j] += h2[0] * W3[j] + h2[1] * W3[40 + j] + b3[j];
}

__global__ void k_mlp_nv(const float* __restrict__ hoF, const float* __restrict__ hoB,
                         MlpP mp, const float* __restrict__ emb1,
                         const float* __restrict__ emb2,
                         _Float16* __restrict__ nv1, _Float16* __restrict__ nv2) {
  int id = blockIdx.x * blockDim.x + threadIdx.x;
  if (id >= B_ * N_) return;
  int n = id & (N_ - 1);
  const float* hF = hoF + (size_t)id * DIN_;
  const float* hB = hoB + (size_t)id * DIN_;
  float f1[40], f2[40];
  for (int j = 0; j < 40; ++j) { f1[j] = 0.f; f2[j] = 0.f; }
  mlp_eval(hF, mp.p[0], f1);  // h1a on A0-chain
  mlp_eval(hB, mp.p[1], f1);  // h1b on A1-chain
  mlp_eval(hF, mp.p[2], f2);  // h2a
  mlp_eval(hB, mp.p[3], f2);  // h2b
  for (int d = 0; d < 40; ++d) {
    nv1[(size_t)id * NDP_ + d] = (_Float16)tanhf(3.0f * emb1[n * 40 + d] * f1[d]);
    nv2[(size_t)id * NDP_ + d] = (_Float16)tanhf(3.0f * emb2[n * 40 + d] * f2[d]);
  }
}

// a = M - M^T is antisymmetric: row v of tanh(3a) yields both prep(adj_d) and
// prep(adj_d^T) columns v of the (pre-transposed) GEMM operands.
__global__ void k_dyn_adj(const float* __restrict__ M, _Float16* __restrict__ AopF,
                          _Float16* __restrict__ AopB) {
  int b = blockIdx.y, v = blockIdx.x, w = threadIdx.x;
  const float* Mb = M + (size_t)b * N_ * N_;
  float t = tanhf(3.0f * (Mb[(size_t)v * N_ + w] - Mb[(size_t)w * N_ + v]));
  float p = t > 0.f ? t : 0.f;
  float q = t < 0.f ? -t : 0.f;
  __shared__ float sp[256], sq[256];
  sp[w] = p; sq[w] = q;
  __syncthreads();
  for (int s = 128; s > 0; s >>= 1) {
    if (w < s) { sp[w] += sp[w + s]; sq[w] += sq[w + s]; }
    __syncthreads();
  }
  float rp = 1.f / (sp[0] + 1.f), rq = 1.f / (sq[0] + 1.f);
  float dg = (v == w) ? 1.f : 0.f;
  size_t o = (size_t)b * N_ * N_ + (size_t)w * N_ + v;
  AopF[o] = (_Float16)((p + dg) * rp);
  AopB[o] = (_Float16)((q + dg) * rq);
}

__global__ void k_gate_zr(const float* __restrict__ zr, const float* __restrict__ bz,
                          const float* __restrict__ br, const float* __restrict__ comb,
                          const float* __restrict__ H, float* __restrict__ zbuf,
                          float* __restrict__ tmpf, _Float16* __restrict__ XT0,
                          _Float16* __restrict__ conc2) {
  int id = blockIdx.x * blockDim.x + threadIdx.x;
  if (id >= B_ * N_) return;
  int n = id & (N_ - 1), b = id >> 8;
  float x = comb[(size_t)id * CH_];
  tmpf[(size_t)id * CH_] = x;
  conc2[(size_t)id * KCAT_] = (_Float16)x;
  conc2[(size_t)id * KCAT_ + 240] = (_Float16)x;
  XT0[((size_t)b * CHP_) * N_ + n] = (_Float16)x;
  for (int o = 0; o < HID_; ++o) {
    float z = 1.f / (1.f + __expf(-(zr[(size_t)id * 128 + o] + bz[o])));
    float r = 1.f / (1.f + __expf(-(zr[(size_t)id * 128 + 64 + o] + br[o])));
    zbuf[(size_t)id * HID_ + o] = z;
    float tv = r * H[(size_t)id * HID_ + o];
    tmpf[(size_t)id * CH_ + 1 + o] = tv;
    conc2[(size_t)id * KCAT_ + 1 + o] = (_Float16)tv;
    conc2[(size_t)id * KCAT_ + 241 + o] = (_Float16)tv;
    XT0[((size_t)b * CHP_ + 1 + o) * N_ + n] = (_Float16)tv;
  }
}

__global__ void k_gru(const float* __restrict__ Cc, const float* __restrict__ bc,
                      const float* __restrict__ zbuf, float* __restrict__ H,
                      const float* __restrict__ wf, const float* __restrict__ bf,
                      float* __restrict__ dout, float* __restrict__ di, int t) {
  int id = blockIdx.x * blockDim.x + threadIdx.x;
  if (id >= B_ * N_) return;
  float acc = 0.f;
  for (int o = 0; o < HID_; ++o) {
    float C = tanhf(Cc[(size_t)id * HID_ + o] + bc[o]);
    float z = zbuf[(size_t)id * HID_ + o];
    float h = z * H[(size_t)id * HID_ + o] + (1.f - z) * C;
    H[(size_t)id * HID_ + o] = h;
    acc += h * wf[o];
  }
  if (dout) {
    float v = acc + bf[0];
    dout[(size_t)id * T_ + t] = v;  // (B,N,T,1) layout
    di[id] = v;
  }
}

__global__ void k_pack_adj(const float* __restrict__ adj, _Float16* AhF, _Float16* AhB) {
  int e = blockIdx.x * blockDim.x + threadIdx.x;
  if (e >= N_ * N_) return;
  int w = e >> 8, v = e & (N_ - 1);
  AhF[e] = (_Float16)adj[(size_t)v * N_ + w];  // A-op for conv with A0 = adj
  AhB[e] = (_Float16)adj[e];                   // A-op for conv with A1 = adj^T
}

// Pack gate weights transposed+padded: WzrT (128x480: z|r), WcT (64x480).
__global__ void k_pack_wt(const float* zW1, const float* zW2, const float* rW1,
                          const float* rW2, const float* cW1, const float* cW2,
                          _Float16* WzrT, _Float16* WcT) {
  int e = blockIdx.x * blockDim.x + threadIdx.x;
  if (e >= 192 * KCAT_) return;
  int row = e / KCAT_, k = e % KCAT_;
  int half = k / 240, kk = k % 240, seg = kk / CHP_, cc = kk % CHP_;
  float val = 0.f;
  if (cc < CH_) {
    int i = seg * CH_ + cc;
    const float* W;
    int oc;
    if (row < 64)        { W = half ? zW2 : zW1; oc = row; }
    else if (row < 128)  { W = half ? rW2 : rW1; oc = row - 64; }
    else                 { W = half ? cW2 : cW1; oc = row - 128; }
    val = W[i * 64 + oc];
  }
  if (row < 128) WzrT[(size_t)row * KCAT_ + k] = (_Float16)val;
  else           WcT[(size_t)(row - 128) * KCAT_ + k] = (_Float16)val;
}

__global__ void k_pack_bias(const float* zb1, const float* zb2, const float* rb1,
                            const float* rb2, const float* cb1, const float* cb2,
                            float* bz, float* br, float* bc) {
  int o = threadIdx.x;
  if (o < 64) { bz[o] = zb1[o] + zb2[o]; br[o] = rb1[o] + rb2[o]; bc[o] = cb1[o] + cb2[o]; }
}

// ---------------------------------------------------------------------------
struct Lin { const float* W; const float* b; };
struct Mlp { Lin fc1, fc2, fc3; };
struct Branch { Lin gc1, gc2, gr1, gr2, gz1, gz2; Mlp h1a, h1b, h2a, h2b; };

extern "C" void kernel_launch(void* const* d_in, const int* in_sizes, int n_in,
                              void* d_out, int out_size, void* d_ws, size_t ws_size,
                              hipStream_t stream) {
  (void)in_sizes; (void)n_in; (void)out_size; (void)ws_size;
  // Inputs flattened in jax pytree order (dict keys sorted alphabetically):
  // input, adj, params{dec, emb1, emb2, enc, fc_final}, task_level.
  // branch keys: gc1,gc2,gr1,gr2,gz1,gz2,h1a,h1b,h2a,h2b ; lin=(W,b) ; mlp: fc1,fc2,fc3
  int ii = 0;
  const float* input = (const float*)d_in[ii++];
  const float* adj   = (const float*)d_in[ii++];
  auto getLin = [&]() { Lin l; l.W = (const float*)d_in[ii++]; l.b = (const float*)d_in[ii++]; return l; };
  auto getMlp = [&]() { Mlp m; m.fc1 = getLin(); m.fc2 = getLin(); m.fc3 = getLin(); return m; };
  auto getBranch = [&]() {
    Branch b;
    b.gc1 = getLin(); b.gc2 = getLin(); b.gr1 = getLin(); b.gr2 = getLin();
    b.gz1 = getLin(); b.gz2 = getLin();
    b.h1a = getMlp(); b.h1b = getMlp(); b.h2a = getMlp(); b.h2b = getMlp();
    return b;
  };
  Branch dec = getBranch();
  const float* emb1 = (const float*)d_in[ii++];
  const float* emb2 = (const float*)d_in[ii++];
  Branch enc = getBranch();
  Lin fcf = getLin();
  // task_level (d_in[ii]) == T_ by construction; hard-wired.

  // ---- workspace carve-up ----
  size_t off = 0;
  auto alloc = [&](size_t bytes) -> char* {
    char* p = (char*)d_ws + off;
    off += (bytes + 255) & ~(size_t)255;
    return p;
  };
  _Float16* AhF = (_Float16*)alloc((size_t)N_ * N_ * 2);
  _Float16* AhB = (_Float16*)alloc((size_t)N_ * N_ * 2);
  _Float16* WzrT_e = (_Float16*)alloc((size_t)128 * KCAT_ * 2);
  _Float16* WcT_e  = (_Float16*)alloc((size_t)64 * KCAT_ * 2);
  _Float16* WzrT_d = (_Float16*)alloc((size_t)128 * KCAT_ * 2);
  _Float16* WcT_d  = (_Float16*)alloc((size_t)64 * KCAT_ * 2);
  float* bz_e = (float*)alloc(64 * 4); float* br_e = (float*)alloc(64 * 4); float* bc_e = (float*)alloc(64 * 4);
  float* bz_d = (float*)alloc(64 * 4); float* br_d = (float*)alloc(64 * 4); float* bc_d = (float*)alloc(64 * 4);
  float* H    = (float*)alloc((size_t)B_ * N_ * HID_ * 4);
  float* di   = (float*)alloc((size_t)B_ * N_ * 4);
  float* comb = (float*)alloc((size_t)B_ * N_ * CH_ * 4);
  float* tmpf = (float*)alloc((size_t)B_ * N_ * CH_ * 4);
  _Float16* XT0 = (_Float16*)alloc((size_t)B_ * CHP_ * N_ * 2);
  _Float16* XT1 = (_Float16*)alloc((size_t)B_ * CHP_ * N_ * 2);
  float* Yg  = (float*)alloc((size_t)B_ * N_ * CHP_ * 4);
  float* hoF = (float*)alloc((size_t)B_ * N_ * DIN_ * 4);
  float* hoB = (float*)alloc((size_t)B_ * N_ * DIN_ * 4);
  _Float16* nv1 = (_Float16*)alloc((size_t)B_ * N_ * NDP_ * 2);
  _Float16* nv2 = (_Float16*)alloc((size_t)B_ * N_ * NDP_ * 2);
  float* Mmat = (float*)alloc((size_t)B_ * N_ * N_ * 4);
  _Float16* AopF = (_Float16*)alloc((size_t)B_ * N_ * N_ * 2);
  _Float16* AopB = (_Float16*)alloc((size_t)B_ * N_ * N_ * 2);
  _Float16* conc  = (_Float16*)alloc((size_t)B_ * N_ * KCAT_ * 2);
  _Float16* conc2 = (_Float16*)alloc((size_t)B_ * N_ * KCAT_ * 2);
  float* zr   = (float*)alloc((size_t)B_ * N_ * 128 * 4);
  float* zbuf = (float*)alloc((size_t)B_ * N_ * HID_ * 4);
  float* Cc   = (float*)alloc((size_t)B_ * N_ * HID_ * 4);

  // zero: state + every padded f16 buffer (pads stay zero forever after)
  hipMemsetAsync(H, 0, (size_t)B_ * N_ * HID_ * 4, stream);
  hipMemsetAsync(di, 0, (size_t)B_ * N_ * 4, stream);
  hipMemsetAsync(XT0, 0, (size_t)B_ * CHP_ * N_ * 2, stream);
  hipMemsetAsync(XT1, 0, (size_t)B_ * CHP_ * N_ * 2, stream);
  hipMemsetAsync(conc, 0, (size_t)B_ * N_ * KCAT_ * 2, stream);
  hipMemsetAsync(conc2, 0, (size_t)B_ * N_ * KCAT_ * 2, stream);
  hipMemsetAsync(nv1, 0, (size_t)B_ * N_ * NDP_ * 2, stream);
  hipMemsetAsync(nv2, 0, (size_t)B_ * N_ * NDP_ * 2, stream);

  // constant packing (recomputed every call; deterministic)
  k_pack_adj<<<(N_ * N_ + 255) / 256, 256, 0, stream>>>(adj, AhF, AhB);
  int wtBlocks = (192 * KCAT_ + 255) / 256;
  k_pack_wt<<<wtBlocks, 256, 0, stream>>>(enc.gz1.W, enc.gz2.W, enc.gr1.W, enc.gr2.W,
                                          enc.gc1.W, enc.gc2.W, WzrT_e, WcT_e);
  k_pack_wt<<<wtBlocks, 256, 0, stream>>>(dec.gz1.W, dec.gz2.W, dec.gr1.W, dec.gr2.W,
                                          dec.gc1.W, dec.gc2.W, WzrT_d, WcT_d);
  k_pack_bias<<<1, 64, 0, stream>>>(enc.gz1.b, enc.gz2.b, enc.gr1.b, enc.gr2.b,
                                    enc.gc1.b, enc.gc2.b, bz_e, br_e, bc_e);
  k_pack_bias<<<1, 64, 0, stream>>>(dec.gz1.b, dec.gz2.b, dec.gr1.b, dec.gr2.b,
                                    dec.gc1.b, dec.gc2.b, bz_d, br_d, bc_d);

  auto mkMlpP = [&](const Branch& P) {
    MlpP m;
    const Mlp* ms[4] = {&P.h1a, &P.h1b, &P.h2a, &P.h2b};
    for (int i = 0; i < 4; ++i) {
      m.p[i][0] = ms[i]->fc1.W; m.p[i][1] = ms[i]->fc1.b;
      m.p[i][2] = ms[i]->fc2.W; m.p[i][3] = ms[i]->fc2.b;
      m.p[i][4] = ms[i]->fc3.W; m.p[i][5] = ms[i]->fc3.b;
    }
    return m;
  };
  MlpP mp_e = mkMlpP(enc), mp_d = mkMlpP(dec);

  const int CMB = (B_ * N_ * CH_ + 255) / 256;
  dim3 gGraph(2, B_);

  auto step = [&](bool isDec, int t, MlpP& mp, _Float16* WzrT, _Float16* WcT,
                  float* bz, float* br, float* bc) {
    const float* xin = isDec ? di : (input + t);
    int xstride = isDec ? 1 : T_;
    k_pack<<<16, 256, 0, stream>>>(xin, xstride, H, comb, XT0, conc, hoF, hoB);
    // hyper chains (static A only), shared between h{1,2}a / h{1,2}b
    for (int dir = 0; dir < 2; ++dir) {
      const _Float16* Ah = dir ? AhB : AhF;
      float* ho = dir ? hoB : hoF;
      k_gemm_graph<false><<<gGraph, 256, 0, stream>>>(Ah, 0, nullptr, 0, XT0, Yg);
      k_combine<<<CMB, 256, 0, stream>>>(Yg, comb, ho, 65, nullptr, 0, XT1);
      k_gemm_graph<false><<<gGraph, 256, 0, stream>>>(Ah, 0, nullptr, 0, XT1, Yg);
      k_combine<<<CMB, 256, 0, stream>>>(Yg, comb, ho, 130, nullptr, 0, nullptr);
    }
    k_mlp_nv<<<16, 256, 0, stream>>>(hoF, hoB, mp, emb1, emb2, nv1, nv2);
    // gram matrix M = nv1 @ nv2^T (K padded 40->64 with zeros)
    k_gemm_nt4<NDP_><<<dim3(2, 4, B_), 256, 0, stream>>>(nv1, NDP_, (long)N_ * NDP_,
                                                         nv2, NDP_, (long)N_ * NDP_,
                                                         Mmat, N_, (long)N_ * N_);
    k_dyn_adj<<<dim3(N_, B_), 256, 0, stream>>>(Mmat, AopF, AopB);
    // RNN chains: dual-A gemm fuses dynamic + static adjacency (both * 0.95)
    auto rnnChains = [&](_Float16* cb, const float* resid) {
      for (int dir = 0; dir < 2; ++dir) {
        const _Float16* Ad = dir ? AopB : AopF;
        const _Float16* Ah = dir ? AhB : AhF;
        int base = dir ? 240 : 0;
        k_gemm_graph<true><<<gGraph, 256, 0, stream>>>(Ad, (long)N_ * N_, Ah, 0, XT0, Yg);
        k_combine<<<CMB, 256, 0, stream>>>(Yg, resid, nullptr, 0, cb, base + 80, XT1);
        k_gemm_graph<true><<<gGraph, 256, 0, stream>>>(Ad, (long)N_ * N_, Ah, 0, XT1, Yg);
        k_combine<<<CMB, 256, 0, stream>>>(Yg, resid, nullptr, 0, cb, base + 160, nullptr);
      }
    };
    rnnChains(conc, comb);
    // z|r merged: (B*N x 480) @ (480 x 128)
    k_gemm_nt4<KCAT_><<<dim3(32, 2, 1), 256, 0, stream>>>(conc, KCAT_, 0, WzrT, KCAT_, 0,
                                                          zr, 128, 0);
    k_gate_zr<<<16, 256, 0, stream>>>(zr, bz, br, comb, H, zbuf, tmpf, XT0, conc2);
    rnnChains(conc2, tmpf);
    k_gemm_nt4<KCAT_><<<dim3(32, 1, 1), 256, 0, stream>>>(conc2, KCAT_, 0, WcT, KCAT_, 0,
                                                          Cc, HID_, 0);
    k_gru<<<16, 256, 0, stream>>>(Cc, bc, zbuf, H, fcf.W, fcf.b,
                                  isDec ? (float*)d_out : nullptr,
                                  isDec ? di : nullptr, t);
  };

  for (int t = 0; t < T_; ++t) step(false, t, mp_e, WzrT_e, WcT_e, bz_e, br_e, bc_e);
  for (int t = 0; t < T_; ++t) step(true, t, mp_d, WzrT_d, WcT_d, bz_d, br_d, bc_d);
}
